// Lumina2TransformerBlock_70841190580228
// MI455X (gfx1250) — matrix operand order repaired and checked by reading, not compiled, not run on hardware
//
#include <hip/hip_runtime.h>
#include <hip/hip_bf16.h>

// ---------------------------------------------------------------------------
// Lumina2 transformer block for MI455X (gfx1250, wave32, WMMA + TDM).
// GEMMs run on v_wmma_f32_16x16x32_bf16 (f32 accum). Activations that feed
// GEMM A-operands are produced in bf16 so the Tensor Data Mover can stage
// A-tiles into LDS (tensor_load_to_lds + s_wait_tensorcnt). K-loops are
// double-buffered: tile k+1 DMAs/stages while tile k runs on the WMMA pipe.
// ---------------------------------------------------------------------------

constexpr int kDIM   = 2304;
constexpr int kH     = 24;
constexpr int kKVH   = 8;
constexpr int kHD    = 96;          // kDIM / kH
constexpr int kKVDIM = kKVH * kHD;  // 768
constexpr int kINNER = 6144;
constexpr int kB     = 2;
constexpr int kS     = 2048;
constexpr int kTEMB  = 1024;
constexpr float kEPS   = 1e-5f;
constexpr float kQKEPS = 1e-5f;

typedef __attribute__((ext_vector_type(16))) __bf16 v16bf;
typedef __attribute__((ext_vector_type(8)))  float  v8f;
typedef __attribute__((ext_vector_type(4)))  float  v4f;
typedef __attribute__((ext_vector_type(4)))  unsigned int v4u;
typedef __attribute__((ext_vector_type(2)))  unsigned int v2u;
typedef __attribute__((ext_vector_type(4)))  unsigned int u32x4;
typedef __attribute__((ext_vector_type(8)))  int i32x8;
typedef __attribute__((ext_vector_type(4)))  int i32x4;

#if __has_builtin(__builtin_amdgcn_tensor_load_to_lds)
#define HAVE_TDM 1
#else
#define HAVE_TDM 0
#endif

// fp32 -> bf16 (round-to-nearest-even) as raw u16.
__device__ __forceinline__ unsigned short f2bf(float f) {
  union { float f; unsigned int u; } cv; cv.f = f;
  unsigned int u = cv.u;
  u += 0x7FFFu + ((u >> 16) & 1u);
  return (unsigned short)(u >> 16);
}
__device__ __forceinline__ unsigned int pack2bf(float a, float b) {
  return (unsigned int)f2bf(a) | ((unsigned int)f2bf(b) << 16);
}

__device__ __forceinline__ void wait_tdm() {
#if __has_builtin(__builtin_amdgcn_s_wait_tensorcnt)
  __builtin_amdgcn_s_wait_tensorcnt(0);
#else
  asm volatile("s_wait_tensorcnt 0x0" ::: "memory");
#endif
}

union FragU { v16bf v; v4u u[2]; };

// A-fragment (16x32 bf16): lane m=l%16, kb=(l/16)*8, K chunks [kb..kb+7],
// [kb+16..kb+23] (CDNA5 ISA 7.12.2 layout).
__device__ __forceinline__ v16bf load_frag_a(const unsigned short* base, int stride) {
  const int lane = threadIdx.x & 31;
  const unsigned short* p = base + (lane & 15) * stride + ((lane >> 4) << 3);
  FragU f;
  f.u[0] = *(const v4u*)(p);
  f.u[1] = *(const v4u*)(p + 16);
  return f.v;
}

// B-fragment (32x16 bf16) from N-major (transposed) tile: lane n=l%16,
// kb=(l/16)*8, K chunks [kb..kb+7], [kb+16..kb+23].
__device__ __forceinline__ v16bf load_frag_b(const unsigned short* base, int stride) {
  const int lane = threadIdx.x & 31;
  const unsigned short* p = base + (lane & 15) * stride + ((lane >> 4) << 3);
  FragU f;
  f.u[0] = *(const v4u*)(p);
  f.u[1] = *(const v4u*)(p + 16);
  return f.v;
}

__device__ __forceinline__ v8f zero_v8f() {
  v8f z;
#pragma unroll
  for (int e = 0; e < 8; ++e) z[e] = 0.f;
  return z;
}

// ---------------------------------------------------------------------------
// 1) Modulation: mod[b][n] = silu(temb[b]) @ w_mod[:,n] + b_mod[n]
// ---------------------------------------------------------------------------
__global__ __launch_bounds__(256) void mod_kernel(
    const float* __restrict__ temb, const float* __restrict__ wmod,
    const float* __restrict__ bmod, float* __restrict__ modv) {
  const int n = blockIdx.x * 256 + threadIdx.x;   // < 4*kDIM
  const int b = blockIdx.y;
  float acc = bmod[n];
  for (int k = 0; k < kTEMB; ++k) {
    const float t = temb[b * kTEMB + k];
    const float s = t / (1.f + __expf(-t));
    acc += s * wmod[(size_t)k * (4 * kDIM) + n];
  }
  modv[(size_t)b * 4 * kDIM + n] = acc;
}

// ---------------------------------------------------------------------------
// 2) norm1: xn = rms(x)*norm1_w*(1+scale_msa)  -> bf16 (feeds QKV GEMMs)
// ---------------------------------------------------------------------------
__global__ __launch_bounds__(256) void norm1_kernel(
    const float* __restrict__ x, const float* __restrict__ w,
    const float* __restrict__ modv, unsigned short* __restrict__ out) {
  const int row = blockIdx.x;
  const int b = row / kS;
  const float* xr = x + (size_t)row * kDIM;
  __shared__ float red[8];
  const int t = threadIdx.x;
  float vals[9];
  float ss = 0.f;
#pragma unroll
  for (int i = 0; i < 9; ++i) { vals[i] = xr[t + 256 * i]; ss += vals[i] * vals[i]; }
#pragma unroll
  for (int o = 16; o; o >>= 1) ss += __shfl_xor(ss, o);
  if ((t & 31) == 0) red[t >> 5] = ss;
  __syncthreads();
  float tot = 0.f;
#pragma unroll
  for (int i = 0; i < 8; ++i) tot += red[i];
  const float rinv = rsqrtf(tot / kDIM + kEPS);
#pragma unroll
  for (int i = 0; i < 9; ++i) {
    const int c = t + 256 * i;
    out[(size_t)row * kDIM + c] =
        f2bf(vals[i] * rinv * w[c] * (1.f + modv[(size_t)b * 4 * kDIM + c]));
  }
}

// ---------------------------------------------------------------------------
// 3) WMMA GEMM: C[M,N](f32) = A[M,K](bf16) @ W[K,N](f32->bf16 at staging).
//    Block tile 128x128, 8 waves x (64x32). Double-buffered LDS: the TDM
//    DMAs A-tile k+1 and all threads stage W-tile k+1 under WMMA of tile k.
// ---------------------------------------------------------------------------
__device__ __forceinline__ void gemm_stage(
    unsigned short (*__restrict__ Asb)[40], unsigned short (*__restrict__ Bsb)[40],
    const unsigned short* __restrict__ A, const float* __restrict__ W,
    int mBlock, int nBlock, int k0, int M, int N, int K, int tid) {
#if HAVE_TDM
  // Tensor Data Mover: DMA the 128x32 bf16 A-tile into LDS. 2D descriptor,
  // tile_dim0=32 elems (64B rows), tile_dim1=128, tensor_dim0_stride=K.
  // pad_interval=16 DW / pad_amount=4 DW reproduces the 80B padded LDS row.
  if (tid < 32) {
    const unsigned lds = (unsigned)(unsigned long long)(const void*)&Asb[0][0];
    const unsigned long long ga =
        (unsigned long long)(const void*)(A + (size_t)mBlock * K + k0);
    u32x4 g0;
    g0[0] = 1u;                                   // count=1, user descriptor
    g0[1] = lds;                                  // lds_addr
    g0[2] = (unsigned)ga;                         // global_addr[31:0]
    g0[3] = (unsigned)(ga >> 32) | (2u << 30);    // global_addr[56:32] | type=2
    i32x8 g1;
    g1[0] = (1 << 16) | (1 << 20) | (3 << 22) | (3 << 25); // 2B data, pad 16DW->+4DW
    g1[1] = (int)((K & 0xffff) << 16);                          // tensor_dim0 lo
    g1[2] = (int)(((unsigned)K >> 16) | (((unsigned)M & 0xffff) << 16)); // d0 hi|d1 lo
    g1[3] = (int)((((unsigned)M >> 16) & 0xffff) | (32u << 16));         // d1 hi|tile_dim0
    g1[4] = 128;                                  // tile_dim1=128, tile_dim2=0
    g1[5] = K;                                    // tensor_dim0_stride lo32
    g1[6] = 0;
    g1[7] = 0;
    i32x4 z4;
#pragma unroll
    for (int e = 0; e < 4; ++e) z4[e] = 0;
#if defined(__clang_major__) && __clang_major__ >= 23
    i32x8 z8;
#pragma unroll
    for (int e = 0; e < 8; ++e) z8[e] = 0;
    __builtin_amdgcn_tensor_load_to_lds(g0, g1, z4, z4, z8, 0);
#else
    __builtin_amdgcn_tensor_load_to_lds(g0, g1, z4, z4, 0);
#endif
  }
#else
  // Fallback: vectorized bf16 copy, 16B per thread-chunk.
#pragma unroll
  for (int i = 0; i < 2; ++i) {
    const int idx = tid + 256 * i;     // 512 chunks of 8 ushorts
    const int r  = idx >> 2;
    const int c8 = (idx & 3) * 8;
    *(v4u*)&Asb[r][c8] = *(const v4u*)&A[(size_t)(mBlock + r) * K + k0 + c8];
  }
#endif
  // Stage W tile 32x128 (f32 -> bf16), transposed into [n][k].
#pragma unroll
  for (int i = 0; i < 2; ++i) {
    const int idx = tid + 256 * i;     // 512 chunks of 8 floats
    const int kk = idx >> 4;           // 0..31
    const int n8 = (idx & 15) * 8;     // 0..120
    const float* wp = &W[(size_t)(k0 + kk) * N + nBlock + n8];
    const v4f w0 = *(const v4f*)(wp);
    const v4f w1v = *(const v4f*)(wp + 4);
#pragma unroll
    for (int j = 0; j < 4; ++j) Bsb[n8 + j][kk] = f2bf(w0[j]);
#pragma unroll
    for (int j = 0; j < 4; ++j) Bsb[n8 + 4 + j][kk] = f2bf(w1v[j]);
  }
}

__global__ __launch_bounds__(256) void gemm_bf16_wmma(
    const unsigned short* __restrict__ A, const float* __restrict__ W,
    float* __restrict__ C, int M, int N, int K) {
  __shared__ __align__(16) unsigned short As[2][128][40];  // [m][k], padded rows
  __shared__ __align__(16) unsigned short Bs[2][128][40];  // [n][k] transposed
  const int tid = threadIdx.x;
  const int nBlock = blockIdx.x * 128;
  const int mBlock = blockIdx.y * 128;
  const int w  = tid >> 5;
  const int wm = w & 1;    // 2 wave-rows  -> 64 M each
  const int wn = w >> 1;   // 4 wave-cols  -> 32 N each

  v8f acc[4][2];
#pragma unroll
  for (int mi = 0; mi < 4; ++mi)
#pragma unroll
    for (int ni = 0; ni < 2; ++ni) acc[mi][ni] = zero_v8f();

  // Prologue: stage tile 0.
  gemm_stage(As[0], Bs[0], A, W, mBlock, nBlock, 0, M, N, K, tid);
#if HAVE_TDM
  if (tid < 32) wait_tdm();
#endif
  __syncthreads();

  const int steps = K / 32;
  for (int s = 0; s < steps; ++s) {
    const int cur = s & 1;
    // Stage next tile into the other buffer while computing this one.
    if (s + 1 < steps)
      gemm_stage(As[cur ^ 1], Bs[cur ^ 1], A, W, mBlock, nBlock,
                 (s + 1) * 32, M, N, K, tid);

    v16bf af[4];
#pragma unroll
    for (int mi = 0; mi < 4; ++mi)
      af[mi] = load_frag_a(&As[cur][wm * 64 + mi * 16][0], 40);
#pragma unroll
    for (int ni = 0; ni < 2; ++ni) {
      const v16bf bf = load_frag_b(&Bs[cur][wn * 32 + ni * 16][0], 40);
#pragma unroll
      for (int mi = 0; mi < 4; ++mi) {
        acc[mi][ni] = __builtin_amdgcn_wmma_f32_16x16x32_bf16(
            false, af[mi], false, bf, (short)0, acc[mi][ni], false, false);
      }
    }
#if HAVE_TDM
    if (s + 1 < steps && tid < 32) wait_tdm();  // next-tile DMA done pre-barrier
#endif
    __syncthreads();
  }

  const int lane = tid & 31;
  const int half = lane >> 4;
#pragma unroll
  for (int mi = 0; mi < 4; ++mi)
#pragma unroll
    for (int ni = 0; ni < 2; ++ni) {
      const int nIdx = nBlock + wn * 32 + ni * 16 + (lane & 15);
#pragma unroll
      for (int v = 0; v < 8; ++v) {
        const int mIdx = mBlock + wm * 64 + mi * 16 + v + 8 * half;
        C[(size_t)mIdx * N + nIdx] = acc[mi][ni][v];
      }
    }
}

// ---------------------------------------------------------------------------
// 4) Per-head RMS norm + RoPE, in place on fp32 q/k. One block per (b,s,h).
// ---------------------------------------------------------------------------
__global__ __launch_bounds__(128) void qknorm_rope_kernel(
    float* __restrict__ x, const float* __restrict__ w,
    const float* __restrict__ cosb, const float* __restrict__ sinb, int heads) {
  const int idx = blockIdx.x;
  const int h  = idx % heads;
  const int bs = idx / heads;
  const int s  = bs % kS;
  float* row = x + ((size_t)bs * heads + h) * kHD;
  __shared__ float sh[kHD];
  __shared__ float red[4];
  const int t = threadIdx.x;
  const float v = (t < kHD) ? row[t] : 0.f;
  float ss = v * v;
#pragma unroll
  for (int o = 16; o; o >>= 1) ss += __shfl_xor(ss, o);
  if ((t & 31) == 0) red[t >> 5] = ss;
  __syncthreads();
  const float tot = red[0] + red[1] + red[2] + red[3];
  const float r = rsqrtf(tot / kHD + kQKEPS);
  if (t < kHD) sh[t] = v * r * w[t];
  __syncthreads();
  if (t < kHD / 2) {
    const float x0 = sh[2 * t], x1 = sh[2 * t + 1];
    const float c  = cosb[(size_t)s * (kHD / 2) + t];
    const float sn = sinb[(size_t)s * (kHD / 2) + t];
    row[2 * t]     = x0 * c - x1 * sn;
    row[2 * t + 1] = x0 * sn + x1 * c;
  }
}

// ---------------------------------------------------------------------------
// 5) Flash attention (GQA 3:1). 4 waves x 16 q rows; WMMA for QK^T and P*V;
//    online softmax; K/V chunks double-buffered. Output in bf16.
// ---------------------------------------------------------------------------
__device__ __forceinline__ void flash_stage(
    unsigned short (*__restrict__ Ksb)[104], unsigned short (*__restrict__ Vtb)[40],
    float* __restrict__ maskbb, const float* __restrict__ Kg,
    const float* __restrict__ Vg, const unsigned char* __restrict__ maskg,
    int b, int hk, int kv0, int tid) {
#pragma unroll
  for (int i = 0; i < 6; ++i) {
    const int idx = tid + 128 * i;    // 768 float4 chunks
    const int r = idx / 24, c4 = (idx % 24) * 4;
    const size_t base = ((size_t)(b * kS + kv0 + r) * kKVH + hk) * kHD + c4;
    const v4f k4 = *(const v4f*)&Kg[base];
    v2u p;
    p[0] = pack2bf(k4[0], k4[1]);
    p[1] = pack2bf(k4[2], k4[3]);
    *(v2u*)&Ksb[r][c4] = p;
    const v4f v4 = *(const v4f*)&Vg[base];
#pragma unroll
    for (int j = 0; j < 4; ++j) Vtb[c4 + j][r] = f2bf(v4[j]);
  }
  if (tid < 32) maskbb[tid] = maskg[b * kS + kv0 + tid] ? 0.f : -1e30f;
}

__global__ __launch_bounds__(128) void flash_attn_kernel(
    const float* __restrict__ Q, const float* __restrict__ Kg,
    const float* __restrict__ Vg, const unsigned char* __restrict__ maskg,
    unsigned short* __restrict__ O, float scale) {
  __shared__ __align__(16) unsigned short Qs[64][104];       // [q][d] (scaled)
  __shared__ __align__(16) unsigned short Ks[2][32][104];    // [kv][d]
  __shared__ __align__(16) unsigned short Vt[2][kHD][40];    // [d][kv]
  __shared__ __align__(16) unsigned short Ps[4][16][32];     // per-wave P
  __shared__ float maskb[2][32];

  const int tid  = threadIdx.x;
  const int lane = tid & 31;
  const int w    = tid >> 5;
  const int half = lane >> 4;
  const int qBlock = blockIdx.x * 64;
  const int bh = blockIdx.y;
  const int b  = bh / kH;
  const int h  = bh % kH;
  const int hk = h / (kH / kKVH);

  // Stage Q tile (softmax scale folded in), float4 loads, packed bf16 stores.
#pragma unroll
  for (int i = 0; i < 12; ++i) {
    const int idx = tid + 128 * i;      // 1536 float4 chunks
    const int r = idx / 24, c4 = (idx % 24) * 4;
    const v4f q4 = *(const v4f*)&Q[((size_t)(b * kS + qBlock + r) * kH + h) * kHD + c4];
    v2u p;
    p[0] = pack2bf(q4[0] * scale, q4[1] * scale);
    p[1] = pack2bf(q4[2] * scale, q4[3] * scale);
    *(v2u*)&Qs[r][c4] = p;
  }

  float mrow[8], lrow[8];
  v8f o[6];
#pragma unroll
  for (int v = 0; v < 8; ++v) { mrow[v] = -1e30f; lrow[v] = 0.f; }
#pragma unroll
  for (int j = 0; j < 6; ++j) o[j] = zero_v8f();

  // Prologue: stage chunk 0.
  flash_stage(Ks[0], Vt[0], maskb[0], Kg, Vg, maskg, b, hk, 0, tid);
  __syncthreads();

  const int chunks = kS / 32;
  for (int c = 0; c < chunks; ++c) {
    const int cur = c & 1;
    if (c + 1 < chunks)
      flash_stage(Ks[cur ^ 1], Vt[cur ^ 1], maskb[cur ^ 1], Kg, Vg, maskg,
                  b, hk, (c + 1) * 32, tid);

    // S = Q K^T: 2 n-tiles of 16 kv cols, K-dim 96 = 3 WMMA steps each.
    v8f sfr[2];
#pragma unroll
    for (int nt = 0; nt < 2; ++nt) {
      sfr[nt] = zero_v8f();
#pragma unroll
      for (int dc = 0; dc < 3; ++dc) {
        const v16bf a  = load_frag_a(&Qs[w * 16][dc * 32], 104);
        const v16bf bb = load_frag_b(&Ks[cur][nt * 16][dc * 32], 104);
        sfr[nt] = __builtin_amdgcn_wmma_f32_16x16x32_bf16(
            false, a, false, bb, (short)0, sfr[nt], false, false);
      }
      const float mb = maskb[cur][nt * 16 + (lane & 15)];
#pragma unroll
      for (int v = 0; v < 8; ++v) sfr[nt][v] += mb;
    }

    // Online softmax update.
    float alpha[8];
#pragma unroll
    for (int v = 0; v < 8; ++v) {
      float mx = fmaxf(sfr[0][v], sfr[1][v]);
#pragma unroll
      for (int o2 = 8; o2; o2 >>= 1) mx = fmaxf(mx, __shfl_xor(mx, o2));
      const float mnew = fmaxf(mrow[v], mx);
      alpha[v] = __expf(mrow[v] - mnew);
      mrow[v] = mnew;
      const float p0 = __expf(sfr[0][v] - mnew);
      const float p1 = __expf(sfr[1][v] - mnew);
      sfr[0][v] = p0; sfr[1][v] = p1;
      float rs = p0 + p1;
#pragma unroll
      for (int o2 = 8; o2; o2 >>= 1) rs += __shfl_xor(rs, o2);
      lrow[v] = lrow[v] * alpha[v] + rs;
    }

    // C-layout P -> LDS -> A-layout fragment.
#pragma unroll
    for (int nt = 0; nt < 2; ++nt)
#pragma unroll
      for (int v = 0; v < 8; ++v)
        Ps[w][v + 8 * half][nt * 16 + (lane & 15)] = f2bf(sfr[nt][v]);
    __syncthreads();
    const v16bf pf = load_frag_a(&Ps[w][0][0], 32);

    // O = O*alpha + P @ V
#pragma unroll
    for (int j = 0; j < 6; ++j) {
#pragma unroll
      for (int v = 0; v < 8; ++v) o[j][v] *= alpha[v];
      const v16bf vf = load_frag_b(&Vt[cur][j * 16][0], 40);
      o[j] = __builtin_amdgcn_wmma_f32_16x16x32_bf16(
          false, pf, false, vf, (short)0, o[j], false, false);
    }
    __syncthreads();
  }

  // Normalize and write bf16 (b, s, h*HD + d): feeds the WO GEMM directly.
#pragma unroll
  for (int j = 0; j < 6; ++j)
#pragma unroll
    for (int v = 0; v < 8; ++v) {
      const int r = w * 16 + v + 8 * half;
      const int d = j * 16 + (lane & 15);
      O[(size_t)(b * kS + qBlock + r) * kDIM + h * kHD + d] = f2bf(o[j][v] / lrow[v]);
    }
}

// ---------------------------------------------------------------------------
// 6) Residual 1 + FFN pre-norm:
//    x(f32) = hidden + tanh(gate_msa)*rms(attn_proj)*norm2_w
//    h(bf16) = rms(x)*ffn_norm1_w*(1+scale_mlp)
// ---------------------------------------------------------------------------
__global__ __launch_bounds__(256) void residual1_kernel(
    const float* __restrict__ hidden, const float* __restrict__ aproj,
    const float* __restrict__ norm2_w, const float* __restrict__ ffn1_w,
    const float* __restrict__ modv, float* __restrict__ x,
    unsigned short* __restrict__ h) {
  const int row = blockIdx.x;
  const int b = row / kS;
  const float* ar = aproj + (size_t)row * kDIM;
  const float* hr = hidden + (size_t)row * kDIM;
  __shared__ float red[8];
  const int t = threadIdx.x;
  float av[9];
  float ss = 0.f;
#pragma unroll
  for (int i = 0; i < 9; ++i) { av[i] = ar[t + 256 * i]; ss += av[i] * av[i]; }
#pragma unroll
  for (int o = 16; o; o >>= 1) ss += __shfl_xor(ss, o);
  if ((t & 31) == 0) red[t >> 5] = ss;
  __syncthreads();
  float tot = 0.f;
#pragma unroll
  for (int i = 0; i < 8; ++i) tot += red[i];
  const float r1 = rsqrtf(tot / kDIM + kEPS);
  __syncthreads();

  float xv[9];
  float ss2 = 0.f;
#pragma unroll
  for (int i = 0; i < 9; ++i) {
    const int c = t + 256 * i;
    const float g = tanhf(modv[(size_t)b * 4 * kDIM + kDIM + c]);
    const float val = hr[c] + g * (av[i] * r1 * norm2_w[c]);
    xv[i] = val; ss2 += val * val;
    x[(size_t)row * kDIM + c] = val;
  }
#pragma unroll
  for (int o = 16; o; o >>= 1) ss2 += __shfl_xor(ss2, o);
  if ((t & 31) == 0) red[t >> 5] = ss2;
  __syncthreads();
  float tot2 = 0.f;
#pragma unroll
  for (int i = 0; i < 8; ++i) tot2 += red[i];
  const float r2 = rsqrtf(tot2 / kDIM + kEPS);
#pragma unroll
  for (int i = 0; i < 9; ++i) {
    const int c = t + 256 * i;
    h[(size_t)row * kDIM + c] =
        f2bf(xv[i] * r2 * ffn1_w[c] * (1.f + modv[(size_t)b * 4 * kDIM + 2 * kDIM + c]));
  }
}

// ---------------------------------------------------------------------------
// 7) SwiGLU: g(bf16) = silu(a1) * a3, vectorized x4.
// ---------------------------------------------------------------------------
__global__ __launch_bounds__(256) void swiglu_kernel(
    const float* __restrict__ a1, const float* __restrict__ a3,
    unsigned short* __restrict__ g, size_t n4) {
  const size_t i = (size_t)blockIdx.x * 256 + threadIdx.x;
  if (i < n4) {
    const v4f a = *(const v4f*)(a1 + 4 * i);
    const v4f c = *(const v4f*)(a3 + 4 * i);
    float s[4];
#pragma unroll
    for (int j = 0; j < 4; ++j) s[j] = (a[j] / (1.f + __expf(-a[j]))) * c[j];
    v2u p;
    p[0] = pack2bf(s[0], s[1]);
    p[1] = pack2bf(s[2], s[3]);
    *(v2u*)(g + 4 * i) = p;
  }
}

// ---------------------------------------------------------------------------
// 8) Residual 2: out = x + tanh(gate_mlp)*rms(mlp)*ffn_norm2_w
// ---------------------------------------------------------------------------
__global__ __launch_bounds__(256) void residual2_kernel(
    const float* __restrict__ x, const float* __restrict__ mlp,
    const float* __restrict__ ffn2_w, const float* __restrict__ modv,
    float* __restrict__ out) {
  const int row = blockIdx.x;
  const int b = row / kS;
  const float* mr = mlp + (size_t)row * kDIM;
  __shared__ float red[8];
  const int t = threadIdx.x;
  float mv[9];
  float ss = 0.f;
#pragma unroll
  for (int i = 0; i < 9; ++i) { mv[i] = mr[t + 256 * i]; ss += mv[i] * mv[i]; }
#pragma unroll
  for (int o = 16; o; o >>= 1) ss += __shfl_xor(ss, o);
  if ((t & 31) == 0) red[t >> 5] = ss;
  __syncthreads();
  float tot = 0.f;
#pragma unroll
  for (int i = 0; i < 8; ++i) tot += red[i];
  const float r = rsqrtf(tot / kDIM + kEPS);
#pragma unroll
  for (int i = 0; i < 9; ++i) {
    const int c = t + 256 * i;
    const float g = tanhf(modv[(size_t)b * 4 * kDIM + 3 * kDIM + c]);
    out[(size_t)row * kDIM + c] = x[(size_t)row * kDIM + c] + g * (mv[i] * r * ffn2_w[c]);
  }
}

// ---------------------------------------------------------------------------
// Host orchestration
// ---------------------------------------------------------------------------
extern "C" void kernel_launch(void* const* d_in, const int* in_sizes, int n_in,
                              void* d_out, int out_size, void* d_ws, size_t ws_size,
                              hipStream_t stream) {
  (void)in_sizes; (void)n_in; (void)out_size; (void)ws_size;
  const float* hidden   = (const float*)d_in[0];
  const float* temb     = (const float*)d_in[1];
  const float* rope_cos = (const float*)d_in[2];
  const float* rope_sin = (const float*)d_in[3];
  const float* w_mod    = (const float*)d_in[4];
  const float* b_mod    = (const float*)d_in[5];
  const float* norm1_w  = (const float*)d_in[6];
  const float* wq       = (const float*)d_in[7];
  const float* wk       = (const float*)d_in[8];
  const float* wv       = (const float*)d_in[9];
  const float* norm_q_w = (const float*)d_in[10];
  const float* norm_k_w = (const float*)d_in[11];
  const float* wo       = (const float*)d_in[12];
  const float* norm2_w  = (const float*)d_in[13];
  const float* ffn1_w   = (const float*)d_in[14];
  const float* w1       = (const float*)d_in[15];
  const float* w2       = (const float*)d_in[16];
  const float* w3       = (const float*)d_in[17];
  const float* ffn2_w   = (const float*)d_in[18];
  const unsigned char* amask = (const unsigned char*)d_in[19];
  float* out = (float*)d_out;

  const size_t M = (size_t)kB * kS;  // 4096 tokens
  char* ws = (char*)d_ws;
  size_t off = 0;
  auto take = [&](size_t bytes) -> void* {
    char* p = ws + off;
    off += (bytes + 255) & ~(size_t)255;
    return (void*)p;
  };
  float*          modv  = (float*)take((size_t)kB * 4 * kDIM * 4);
  unsigned short* xn    = (unsigned short*)take(M * kDIM * 2);   // bf16
  float*          qb    = (float*)take(M * kDIM * 4);
  float*          kbuf  = (float*)take(M * kKVDIM * 4);
  float*          vbuf  = (float*)take(M * kKVDIM * 4);
  unsigned short* attnb = (unsigned short*)take(M * kDIM * 2);   // bf16
  float*          aproj = (float*)take(M * kDIM * 4);
  float*          xres  = (float*)take(M * kDIM * 4);
  float*          a1    = (float*)take(M * kINNER * 4);
  float*          a3    = (float*)take(M * kINNER * 4);
  // Aliases over dead regions:
  unsigned short* hbuf = attnb;               // attn dead after WO GEMM
  unsigned short* gbuf = (unsigned short*)qb; // q+k f32 (50.33MB) == g bf16 size
  float*          mlp  = aproj;               // aproj dead after residual1

  // 1) modulation
  mod_kernel<<<dim3(4 * kDIM / 256, kB), 256, 0, stream>>>(temb, w_mod, b_mod, modv);
  // 2) norm1 + modulate -> bf16
  norm1_kernel<<<(int)M, 256, 0, stream>>>(hidden, norm1_w, modv, xn);
  // 3) QKV projections (WMMA, TDM-staged A, double-buffered)
  gemm_bf16_wmma<<<dim3(kDIM / 128, (int)M / 128), 256, 0, stream>>>(xn, wq, qb, (int)M, kDIM, kDIM);
  gemm_bf16_wmma<<<dim3(kKVDIM / 128, (int)M / 128), 256, 0, stream>>>(xn, wk, kbuf, (int)M, kKVDIM, kDIM);
  gemm_bf16_wmma<<<dim3(kKVDIM / 128, (int)M / 128), 256, 0, stream>>>(xn, wv, vbuf, (int)M, kKVDIM, kDIM);
  // 4) per-head RMS + RoPE (in place, f32)
  qknorm_rope_kernel<<<(int)(M * kH), 128, 0, stream>>>(qb, norm_q_w, rope_cos, rope_sin, kH);
  qknorm_rope_kernel<<<(int)(M * kKVH), 128, 0, stream>>>(kbuf, norm_k_w, rope_cos, rope_sin, kKVH);
  // 5) flash attention (WMMA) -> bf16
  const float scale = rsqrtf((float)kHD);
  flash_attn_kernel<<<dim3(kS / 64, kB * kH), 128, 0, stream>>>(qb, kbuf, vbuf, amask, attnb, scale);
  // 6) output projection (WMMA)
  gemm_bf16_wmma<<<dim3(kDIM / 128, (int)M / 128), 256, 0, stream>>>(attnb, wo, aproj, (int)M, kDIM, kDIM);
  // 7) residual 1 + FFN pre-norm (x f32, h bf16)
  residual1_kernel<<<(int)M, 256, 0, stream>>>(hidden, aproj, norm2_w, ffn1_w, modv, xres, hbuf);
  // 8) FFN up projections (WMMA)
  gemm_bf16_wmma<<<dim3(kINNER / 128, (int)M / 128), 256, 0, stream>>>(hbuf, w1, a1, (int)M, kINNER, kDIM);
  gemm_bf16_wmma<<<dim3(kINNER / 128, (int)M / 128), 256, 0, stream>>>(hbuf, w3, a3, (int)M, kINNER, kDIM);
  // 9) SwiGLU -> bf16
  const size_t n4 = M * kINNER / 4;
  swiglu_kernel<<<(int)((n4 + 255) / 256), 256, 0, stream>>>(a1, a3, gbuf, n4);
  // 10) FFN down projection (WMMA)
  gemm_bf16_wmma<<<dim3(kDIM / 128, (int)M / 128), 256, 0, stream>>>(gbuf, w2, mlp, (int)M, kDIM, kINNER);
  // 11) residual 2 -> output
  residual2_kernel<<<(int)M, 256, 0, stream>>>(xres, mlp, ffn2_w, modv, out);
}
